// SAGEGRU_33406255628523
// MI455X (gfx1250) — matrix operand order repaired and checked by reading, not compiled, not run on hardware
//
#include <hip/hip_runtime.h>
#include <hip/hip_bf16.h>
#include <math.h>
#include <stdint.h>

// ---------------------------------------------------------------------------
// SAGE(2-layer, mean-agg) x T steps -> GRU scan -> linear head.  MI455X/gfx1250.
//   * CSR gather aggregation (L2-resident, no scatter atomics)
//   * fused gather + v_wmma_f32_16x16x32_f16 + LayerNorm per SAGE layer
//   * self-feature tiles staged LDS-ward with GLOBAL_LOAD_ASYNC_TO_LDS_B128
//     (ASYNCcnt path) -- builtin confirmed present in this toolchain
// ---------------------------------------------------------------------------

typedef __attribute__((ext_vector_type(16))) _Float16 v16h_t;
typedef __attribute__((ext_vector_type(8)))  float    v8f_t;

static constexpr int kB  = 8;
static constexpr int kT  = 32;
static constexpr int kN  = 10000;
static constexpr int kF  = 16;
static constexpr int kE  = 160000;
static constexpr int kHG = 64;
static constexpr int kHT = 128;
static constexpr int kRowBlocks = (kN + 31) / 32;   // 313

// ------------------------------------------------- async LDS staging helpers
#if defined(__gfx1250__) && __has_builtin(__builtin_amdgcn_global_load_async_to_lds_b128)
#define ASYNC_LDS 1
#else
#define ASYNC_LDS 0
#endif

#if ASYNC_LDS
typedef int v4i_vs __attribute__((vector_size(4 * sizeof(int))));
typedef __attribute__((address_space(1))) v4i_vs* gptr_v4i;
typedef __attribute__((address_space(3))) v4i_vs* lptr_v4i;
#endif

__device__ __forceinline__ void async_copy16(void* lds_dst, const void* gsrc) {
#if ASYNC_LDS
  __builtin_amdgcn_global_load_async_to_lds_b128(
      (gptr_v4i)(uintptr_t)gsrc,
      (lptr_v4i)(uint32_t)(uintptr_t)lds_dst,
      0, 0);
#else
  *(float4*)lds_dst = *(const float4*)gsrc;
#endif
}

__device__ __forceinline__ void wait_async_copies() {
#if ASYNC_LDS
#if __has_builtin(__builtin_amdgcn_s_wait_asynccnt)
  __builtin_amdgcn_s_wait_asynccnt(0);
#else
  asm volatile("s_wait_asynccnt 0" ::: "memory");
#endif
#endif
}

// ---------------------------------------------------------------- utilities
__global__ void k_zero_f(float* __restrict__ p, size_t n) {
  size_t i = (size_t)blockIdx.x * blockDim.x + threadIdx.x;
  size_t stride = (size_t)gridDim.x * blockDim.x;
  for (; i < n; i += stride) p[i] = 0.0f;
}

__global__ void k_zero_i(int* __restrict__ p, size_t n) {
  size_t i = (size_t)blockIdx.x * blockDim.x + threadIdx.x;
  size_t stride = (size_t)gridDim.x * blockDim.x;
  for (; i < n; i += stride) p[i] = 0;
}

// ----------------------------------------------------------- CSR build (dst)
__global__ void k_count(const int* __restrict__ dst, int* __restrict__ deg) {
  int e = blockIdx.x * blockDim.x + threadIdx.x;
  if (e < kE) atomicAdd(&deg[dst[e]], 1);
}

// single-block exclusive scan over deg -> offs[0..N]; also invdeg, cursor=0
__global__ __launch_bounds__(1024) void k_scan(const int* __restrict__ deg,
                                               int* __restrict__ offs,
                                               int* __restrict__ cursor,
                                               float* __restrict__ invdeg) {
  __shared__ int part[1024];
  const int CH = 10;                       // 1024*10 >= kN
  int tid = threadIdx.x;
  int base = tid * CH;
  int s = 0;
  for (int i = 0; i < CH; ++i) { int n = base + i; if (n < kN) s += deg[n]; }
  part[tid] = s;
  __syncthreads();
  for (int off = 1; off < 1024; off <<= 1) {   // Hillis-Steele inclusive scan
    int v = (tid >= off) ? part[tid - off] : 0;
    __syncthreads();
    part[tid] += v;
    __syncthreads();
  }
  int run = (tid == 0) ? 0 : part[tid - 1];
  for (int i = 0; i < CH; ++i) {
    int n = base + i;
    if (n < kN) {
      offs[n] = run;
      int d = deg[n];
      run += d;
      cursor[n] = 0;
      invdeg[n] = 1.0f / fmaxf((float)d, 1.0f);
    }
  }
  if (tid == 1023) offs[kN] = run;         // == kE
}

__global__ void k_fill(const int* __restrict__ src, const int* __restrict__ dst,
                       const int* __restrict__ offs, int* __restrict__ cursor,
                       int* __restrict__ csr_src) {
  int e = blockIdx.x * blockDim.x + threadIdx.x;
  if (e < kE) {
    int d = dst[e];
    int p = atomicAdd(&cursor[d], 1);
    csr_src[offs[d] + p] = src[e];
  }
}

// --------------------------------------------------------------- WMMA frags
// layouts per CDNA5 ISA 7.12.2 (wave32)
__device__ __forceinline__ v16h_t load_a_frag(const _Float16* lds, int ldw,
                                              int row_base, int kchunk, int lane) {
  v16h_t a;
  int r = row_base + (lane & 15);
  int kb = (lane >> 4) << 3;          // 0 or 8
#pragma unroll
  for (int j = 0; j < 16; ++j) {
    int k = ((j >> 3) << 4) + kb + (((j >> 1) & 3) << 1) + (j & 1);
    a[j] = lds[r * ldw + kchunk * 32 + k];
  }
  return a;
}
__device__ __forceinline__ v16h_t load_b_frag(const _Float16* lds, int ldw,
                                              int col_base, int kchunk, int lane) {
  v16h_t bfr;
  int c = col_base + (lane & 15);
  int kb = (lane >> 4) << 3;          // 0 or 8
#pragma unroll
  for (int j = 0; j < 16; ++j) bfr[j] = lds[(kchunk * 32 + ((j >> 3) << 4) + kb + (j & 7)) * ldw + c];
  return bfr;
}

// ------------------------------------------------------------- SAGE layer 1
// h1 = relu(LN(gather_mean(x) @ W1l.T + b1 + x @ W1r.T)); K=32, 1 wmma/tile
__global__ __launch_bounds__(256) void k_sage1(
    const float* __restrict__ x_seq, const int* __restrict__ offs,
    const int* __restrict__ csr_src, const float* __restrict__ invdeg,
    const float* __restrict__ W1l, const float* __restrict__ W1r,
    const float* __restrict__ b1, const float* __restrict__ g1,
    const float* __restrict__ be1,
    float* __restrict__ h1, int t) {
  __shared__ _Float16 lA[32 * 32];
  __shared__ _Float16 lB[32 * 64];
  __shared__ float    lC[32][64];
  __shared__ float    sx[32 * 16];          // staged self features (f32)
  int b = blockIdx.y;
  int r0 = blockIdx.x * 32;
  int tid = threadIdx.x;
  const size_t xbase = ((size_t)b * kT + t) * kN;

  // async-stage x[b,t, r0..r0+31, :] -> sx (32 rows x 4 x 16B)
  if (tid < 128) {
    int rr = tid >> 2, part = tid & 3;
    int n = r0 + rr;
    if (n < kN)
      async_copy16(&sx[rr * 16 + part * 4],
                   x_seq + (xbase + n) * kF + part * 4);
  }
  // weights Bcat = [W1l ; W1r] along K
  for (int i = tid; i < 32 * 64; i += 256) {
    int k = i >> 6, h = i & 63;
    float w = (k < 16) ? W1l[h * 16 + k] : W1r[h * 16 + (k - 16)];
    lB[k * 64 + h] = (_Float16)w;
  }
  // CSR gather-aggregate: 8 threads/row, 2 features/thread
  {
    int r = tid >> 3, q = tid & 7;
    int n = r0 + r;
    float a0 = 0.0f, a1 = 0.0f;
    if (n < kN) {
      int e1 = offs[n + 1];
      for (int e = offs[n]; e < e1; ++e) {
        const float* xp = x_seq + (xbase + csr_src[e]) * kF + 2 * q;
        a0 += xp[0];
        a1 += xp[1];
      }
      float idg = invdeg[n];
      a0 *= idg; a1 *= idg;
    }
    lA[r * 32 + 2 * q]     = (_Float16)a0;
    lA[r * 32 + 2 * q + 1] = (_Float16)a1;
  }
  wait_async_copies();
  __syncthreads();
  // convert staged self features into lA[:, 16..31]
  for (int i = tid; i < 32 * 16; i += 256) {
    int rr = i >> 4;
    int n = r0 + rr;
    lA[rr * 32 + 16 + (i & 15)] = (_Float16)((n < kN) ? sx[i] : 0.0f);
  }
  __syncthreads();

  int wave = tid >> 5, lane = tid & 31;
  int sub = wave >> 2, ct = wave & 3;
  v16h_t af = load_a_frag(lA, 32, sub << 4, 0, lane);
  v16h_t bf = load_b_frag(lB, 64, ct << 4, 0, lane);
  v8f_t  cf = {};
  cf = __builtin_amdgcn_wmma_f32_16x16x32_f16(false, af, false, bf,
                                              (short)0, cf, false, false);
  int crow = (sub << 4) + ((lane >> 4) << 3);
  int ccol = (ct << 4) + (lane & 15);
  float bias = b1[ccol];
#pragma unroll
  for (int i = 0; i < 8; ++i) lC[crow + i][ccol] = cf[i] + bias;
  __syncthreads();

  if (tid < 32) {                      // LayerNorm + ReLU per node row
    int n = r0 + tid;
    float mu = 0.0f;
#pragma unroll
    for (int j = 0; j < 64; ++j) mu += lC[tid][j];
    mu *= (1.0f / 64.0f);
    float var = 0.0f;
#pragma unroll
    for (int j = 0; j < 64; ++j) { float d = lC[tid][j] - mu; var += d * d; }
    var *= (1.0f / 64.0f);
    float rs = rsqrtf(var + 1e-5f);
    if (n < kN) {
      float* op = h1 + ((size_t)b * kN + n) * kHG;
#pragma unroll
      for (int j = 0; j < 64; ++j)
        op[j] = fmaxf((lC[tid][j] - mu) * rs * g1[j] + be1[j], 0.0f);
    }
  }
}

// ------------------------------------------------------------- SAGE layer 2
// h2 = relu(LN(gather_mean(h1) @ W2l.T + b2 + h1 @ W2r.T)); column sums of h2
// accumulated into Hsum[t,b,:].  K=128 -> 4-chained wmma.
__global__ __launch_bounds__(256) void k_sage2(
    const float* __restrict__ h1, const int* __restrict__ offs,
    const int* __restrict__ csr_src, const float* __restrict__ invdeg,
    const float* __restrict__ W2l, const float* __restrict__ W2r,
    const float* __restrict__ b2, const float* __restrict__ g2,
    const float* __restrict__ be2,
    float* __restrict__ Hsum, int t) {
  __shared__ _Float16 lA[32 * 128];   // 8 KB
  __shared__ _Float16 lB[128 * 64];   // 16 KB
  __shared__ float    lC[32][64];     // 8 KB
  __shared__ float    sx[32 * 64];    // 8 KB staged self features
  int b = blockIdx.y;
  int r0 = blockIdx.x * 32;
  int tid = threadIdx.x;
  const float* hb = h1 + (size_t)b * kN * kHG;

  // async-stage h1[b, r0..r0+31, :] -> sx (32 rows x 16 x 16B = 512 chunks)
  for (int idx = tid; idx < 512; idx += 256) {
    int rr = idx >> 4, part = idx & 15;
    int n = r0 + rr;
    if (n < kN)
      async_copy16(&sx[rr * 64 + part * 4], hb + (size_t)n * kHG + part * 4);
  }
  // weights Bcat = [W2l ; W2r]
  for (int i = tid; i < 128 * 64; i += 256) {
    int k = i >> 6, h = i & 63;
    float w = (k < 64) ? W2l[h * 64 + k] : W2r[h * 64 + (k - 64)];
    lB[k * 64 + h] = (_Float16)w;
  }
  // CSR gather-aggregate: 8 threads/row, 8 features/thread
  {
    int r = tid >> 3, q = tid & 7;
    int n = r0 + r;
    float acc[8];
#pragma unroll
    for (int j = 0; j < 8; ++j) acc[j] = 0.0f;
    if (n < kN) {
      int e1 = offs[n + 1];
      for (int e = offs[n]; e < e1; ++e) {
        const float* hp = hb + (size_t)csr_src[e] * kHG + 8 * q;
#pragma unroll
        for (int j = 0; j < 8; ++j) acc[j] += hp[j];
      }
      float idg = invdeg[n];
#pragma unroll
      for (int j = 0; j < 8; ++j) acc[j] *= idg;
    }
#pragma unroll
    for (int j = 0; j < 8; ++j) lA[r * 128 + 8 * q + j] = (_Float16)acc[j];
  }
  wait_async_copies();
  __syncthreads();
  // staged self features -> lA[:, 64..127]
  for (int i = tid; i < 32 * 64; i += 256) {
    int rr = i >> 6;
    int n = r0 + rr;
    lA[rr * 128 + 64 + (i & 63)] = (_Float16)((n < kN) ? sx[i] : 0.0f);
  }
  __syncthreads();

  int wave = tid >> 5, lane = tid & 31;
  int sub = wave >> 2, ct = wave & 3;
  v8f_t cf = {};
#pragma unroll
  for (int kc = 0; kc < 4; ++kc) {
    v16h_t af = load_a_frag(lA, 128, sub << 4, kc, lane);
    v16h_t bf = load_b_frag(lB, 64, ct << 4, kc, lane);
    cf = __builtin_amdgcn_wmma_f32_16x16x32_f16(false, af, false, bf,
                                                (short)0, cf, false, false);
  }
  int crow = (sub << 4) + ((lane >> 4) << 3);
  int ccol = (ct << 4) + (lane & 15);
  float bias = b2[ccol];
#pragma unroll
  for (int i = 0; i < 8; ++i) lC[crow + i][ccol] = cf[i] + bias;
  __syncthreads();

  if (tid < 32) {                      // LN + ReLU; zero padded rows
    int n = r0 + tid;
    float mu = 0.0f;
#pragma unroll
    for (int j = 0; j < 64; ++j) mu += lC[tid][j];
    mu *= (1.0f / 64.0f);
    float var = 0.0f;
#pragma unroll
    for (int j = 0; j < 64; ++j) { float d = lC[tid][j] - mu; var += d * d; }
    var *= (1.0f / 64.0f);
    float rs = rsqrtf(var + 1e-5f);
#pragma unroll
    for (int j = 0; j < 64; ++j) {
      float v = fmaxf((lC[tid][j] - mu) * rs * g2[j] + be2[j], 0.0f);
      lC[tid][j] = (n < kN) ? v : 0.0f;
    }
  }
  __syncthreads();

  if (tid < 64) {                      // block column sum -> Hsum[t,b,:]
    float s = 0.0f;
#pragma unroll
    for (int r = 0; r < 32; ++r) s += lC[r][tid];
    atomicAdd(&Hsum[((size_t)t * kB + b) * kHG + tid], s);
  }
}

// --------------------------------------------------------- GRU scan + head
__global__ __launch_bounds__(1024) void k_gru(
    const float* __restrict__ Hsum,
    const float* __restrict__ W_ih, const float* __restrict__ W_hh,
    const float* __restrict__ b_ih, const float* __restrict__ b_hh,
    const float* __restrict__ Wh, const float* __restrict__ bh,
    float* __restrict__ out) {
  __shared__ float hs[kB][kHT];
  __shared__ float xs[kB][kHG];
  int tid = threadIdx.x;
  int b = tid >> 7, j = tid & 127;
  hs[b][j] = 0.0f;
  __syncthreads();
  for (int t = 0; t < kT; ++t) {
    if (j < kHG)
      xs[b][j] = Hsum[((size_t)t * kB + b) * kHG + j] * (1.0f / (float)kN);
    __syncthreads();
    float gi_r = b_ih[j], gi_z = b_ih[kHT + j], gi_n = b_ih[2 * kHT + j];
#pragma unroll 8
    for (int k = 0; k < kHG; ++k) {
      float xv = xs[b][k];
      gi_r += W_ih[(size_t)j * kHG + k] * xv;
      gi_z += W_ih[(size_t)(kHT + j) * kHG + k] * xv;
      gi_n += W_ih[(size_t)(2 * kHT + j) * kHG + k] * xv;
    }
    float gh_r = b_hh[j], gh_z = b_hh[kHT + j], gh_n = b_hh[2 * kHT + j];
#pragma unroll 8
    for (int k = 0; k < kHT; ++k) {
      float hv = hs[b][k];
      gh_r += W_hh[(size_t)j * kHT + k] * hv;
      gh_z += W_hh[(size_t)(kHT + j) * kHT + k] * hv;
      gh_n += W_hh[(size_t)(2 * kHT + j) * kHT + k] * hv;
    }
    float r = 1.0f / (1.0f + expf(-(gi_r + gh_r)));
    float z = 1.0f / (1.0f + expf(-(gi_z + gh_z)));
    float nn = tanhf(gi_n + r * gh_n);
    float hnew = (1.0f - z) * nn + z * hs[b][j];
    __syncthreads();
    hs[b][j] = hnew;
    __syncthreads();
  }
  if (j == 0) {
    float s = bh[0];
#pragma unroll 8
    for (int k = 0; k < kHT; ++k) s += Wh[k] * hs[b][k];
    out[b] = s;
  }
}

// ------------------------------------------------------------------- launch
extern "C" void kernel_launch(void* const* d_in, const int* in_sizes, int n_in,
                              void* d_out, int out_size, void* d_ws, size_t ws_size,
                              hipStream_t stream) {
  const float* x_seq = (const float*)d_in[0];
  const int*   eidx  = (const int*)d_in[1];
  const float* W1l = (const float*)d_in[2];
  const float* b1  = (const float*)d_in[3];
  const float* W1r = (const float*)d_in[4];
  const float* g1  = (const float*)d_in[5];
  const float* be1 = (const float*)d_in[6];
  const float* W2l = (const float*)d_in[7];
  const float* b2  = (const float*)d_in[8];
  const float* W2r = (const float*)d_in[9];
  const float* g2  = (const float*)d_in[10];
  const float* be2 = (const float*)d_in[11];
  const float* W_ih = (const float*)d_in[12];
  const float* W_hh = (const float*)d_in[13];
  const float* b_ih = (const float*)d_in[14];
  const float* b_hh = (const float*)d_in[15];
  const float* Wh   = (const float*)d_in[16];
  const float* bh   = (const float*)d_in[17];

  const int* esrc = eidx;
  const int* edst = eidx + kE;

  // workspace carve-up (all offsets 256B-aligned)
  char* base = (char*)d_ws;
  int*   deg     = (int*)(base);                         // kN
  int*   offs    = (int*)(base + 0x0000A100);            // kN+1
  int*   cursor  = (int*)(base + 0x00014200);            // kN
  int*   csr_src = (int*)(base + 0x0001E300);            // kE
  float* invdeg  = (float*)(base + 0x000BB000);          // kN
  float* Hsum    = (float*)(base + 0x000C5100);          // kT*kB*kHG
  float* h1      = (float*)(base + 0x000D5200);          // kB*kN*kHG (~20.5MB)

  // ---- CSR build + invdeg (once per launch; deterministic)
  k_zero_i<<<64, 256, 0, stream>>>(deg, (size_t)kN);
  k_count<<<(kE + 255) / 256, 256, 0, stream>>>(edst, deg);
  k_scan<<<1, 1024, 0, stream>>>(deg, offs, cursor, invdeg);
  k_fill<<<(kE + 255) / 256, 256, 0, stream>>>(esrc, edst, offs, cursor, csr_src);
  k_zero_f<<<64, 256, 0, stream>>>(Hsum, (size_t)kT * kB * kHG);

  dim3 sageGrid(kRowBlocks, kB);
  for (int t = 0; t < kT; ++t) {
    k_sage1<<<sageGrid, 256, 0, stream>>>(x_seq, offs, csr_src, invdeg,
                                          W1l, W1r, b1, g1, be1, h1, t);
    k_sage2<<<sageGrid, 256, 0, stream>>>(h1, offs, csr_src, invdeg,
                                          W2l, W2r, b2, g2, be2, Hsum, t);
  }
  k_gru<<<1, 1024, 0, stream>>>(Hsum, W_ih, W_hh, b_ih, b_hh, Wh, bh,
                                (float*)d_out);
}
